// DZSpecimenClfToy_35175782154857
// MI455X (gfx1250) — matrix operand order repaired and checked by reading, not compiled, not run on hardware
//
#include <hip/hip_runtime.h>
#include <hip/hip_bf16.h>

typedef __attribute__((ext_vector_type(16))) _Float16 v16h;
typedef __attribute__((ext_vector_type(8)))  _Float16 v8h;
typedef __attribute__((ext_vector_type(2)))  _Float16 v2h;
typedef __attribute__((ext_vector_type(8)))  float    v8f;

#define B_BATCH   64
#define NPTS      2048
#define K_TOTAL   98304           // 2048 * 48
#define N_OUT     1000
#define H1        4096            // N*2 hidden
#define K_CHUNKS  16
#define K_PER_CHUNK (K_TOTAL / K_CHUNKS)   // 6144
#define K_STEPS   (K_PER_CHUNK / 64)       // 96 steps of K=64 (2 WMMAs each)

// ---------------------------------------------------------------------------
// Kernel 1: 64x64 -> 4x4 bilinear resize (exact 2x2 average), MLP1 + sigmoid,
// map to pixel coords. One block per batch image.
// ---------------------------------------------------------------------------
__global__ void coords_kernel(const float* __restrict__ tv_img,
                              const float* __restrict__ w1,
                              const float* __restrict__ b1,
                              float* __restrict__ coords) {
    __shared__ float tv[48];
    const int b = blockIdx.x;
    const int t = threadIdx.x;
    const float* img = tv_img + (size_t)b * 3 * 64 * 64;
    if (t < 48) {
        int c = t >> 4, rem = t & 15, i = rem >> 2, j = rem & 3;
        const float* p = img + c * 4096;
        int r0 = 7 + 16 * i, c0 = 7 + 16 * j;
        tv[t] = 0.25f * (p[r0 * 64 + c0] + p[r0 * 64 + c0 + 1] +
                         p[(r0 + 1) * 64 + c0] + p[(r0 + 1) * 64 + c0 + 1]);
    }
    __syncthreads();
    for (int jj = t; jj < H1; jj += blockDim.x) {
        float acc = b1[jj];
#pragma unroll
        for (int k = 0; k < 48; ++k) acc += tv[k] * w1[k * H1 + jj];
        float s = 1.f / (1.f + __expf(-acc));
        coords[(size_t)b * H1 + jj] = s * 1019.f + 2.f;   // sig*(H-1-PATCH)+2
    }
}

// ---------------------------------------------------------------------------
// Kernel 2: bilinear 4x4x3 patch gather -> f16 A matrix (row-major, 64 x K).
// One thread per (b, n, i, j); computes 3 channels.
// ---------------------------------------------------------------------------
__global__ void gather_kernel(const float* __restrict__ imgs,
                              const float* __restrict__ coords,
                              _Float16* __restrict__ A) {
    long long tid = (long long)blockIdx.x * blockDim.x + threadIdx.x;
    int pix = (int)(tid & 15);            // i*4 + j
    long long bn = tid >> 4;
    int n = (int)(bn & (NPTS - 1));
    int b = (int)(bn >> 11);
    int i = pix >> 2, j = pix & 3;

    float cx = coords[((size_t)b * NPTS + n) * 2 + 0];
    float cy = coords[((size_t)b * NPTS + n) * 2 + 1];
    float x0 = floorf(cx), y0 = floorf(cy);
    float fx = cx - x0, fy = cy - y0;
    int r = (int)x0 + i - 2;              // in [0, 1022]
    int c = (int)y0 + j - 2;

    const float* img = imgs + (size_t)b * 1024 * 1024 * 3;
    const float* p00 = img + ((size_t)r * 1024 + c) * 3;
    const float* p01 = p00 + 3;
    const float* p10 = p00 + 1024 * 3;
    const float* p11 = p10 + 3;
    float w00 = (1.f - fx) * (1.f - fy);
    float w01 = (1.f - fx) * fy;
    float w10 = fx * (1.f - fy);
    float w11 = fx * fy;

    _Float16* dst = A + (size_t)b * K_TOTAL + (size_t)n * 48 + pix * 3;
#pragma unroll
    for (int ch = 0; ch < 3; ++ch) {
        float v = p00[ch] * w00 + p01[ch] * w01 + p10[ch] * w10 + p11[ch] * w11;
        dst[ch] = (_Float16)v;
    }
}

// ---------------------------------------------------------------------------
// Kernel 3: out[b, o] = b2[o]  (bias init; GEMM accumulates atomically on top)
// ---------------------------------------------------------------------------
__global__ void init_out_kernel(const float* __restrict__ b2,
                                float* __restrict__ out) {
    int t = blockIdx.x * blockDim.x + threadIdx.x;
    if (t < B_BATCH * N_OUT) out[t] = b2[t % N_OUT];
}

// ---------------------------------------------------------------------------
// Kernel 4: WMMA GEMM  out += A(64 x K, f16) * w2(K x 1000, f32->f16)
// Grid: (63 N-tiles) x (K_CHUNKS).  128 threads = 4 waves; wave w owns the
// 16x16 C tile for rows [16w, 16w+16).  w2 streamed once from HBM.
// Staging is software-pipelined: loads for step s+2 issued while the
// (already in-flight) values for step s+1 are converted into the LDS
// double buffer; K=64 per barrier interval (2 WMMAs).
//
// Fragment layouts (ISA 7.12.2):
//   A 16x32 f16: lane-half g -> K in [8g,8g+8) (VGPR0-3) and [16+8g,..) (V4-7)
//   B 32x16 f16: lane-half g -> K in [8g,8g+8) (VGPR0-3) and [16+8g,..) (V4-7)
// ---------------------------------------------------------------------------
__global__ void __launch_bounds__(128)
gemm_kernel(const _Float16* __restrict__ A,
            const float* __restrict__ w2,
            float* __restrict__ out) {
    __shared__ _Float16 Bt[2][16 * 64];   // [n][k] transposed tile, 2x2 KB

    const int nt = blockIdx.x;            // N tile: cols [16*nt, 16*nt+16)
    const int kc = blockIdx.y;            // K chunk
    const int n0 = nt * 16;
    const int kBase = kc * K_PER_CHUNK;

    const int tid  = threadIdx.x;
    const int lane = tid & 31;
    const int wave = tid >> 5;
    const int col  = lane & 15;           // M row (A) / N col (B,C) within tile
    const int g    = lane >> 4;           // half-wave group

    // staging: thread t handles rows {16q+2u, 16q+2u+1 : q=0..3}, column sCol
    const int sCol = tid & 15;
    const int u    = tid >> 4;            // 0..7
    const int nc   = n0 + sCol;
    const float colMask = (nc < N_OUT) ? 1.f : 0.f;   // zero out padded cols
    const int   ncc     = (nc < N_OUT) ? nc : (N_OUT - 1);

    const float* wRow = w2 + (size_t)(kBase + 2 * u) * N_OUT + ncc;
    const size_t stepStride = 64 * (size_t)N_OUT;

    // A fragment base: row = wave*16 + col
    const _Float16* aPtr =
        A + (size_t)(wave * 16 + col) * K_TOTAL + kBase + g * 8;

    v8f acc = {};

    auto stage_load = [&](const float* w, float r[8]) {
#pragma unroll
        for (int q = 0; q < 4; ++q) {
            r[2 * q]     = w[(size_t)(16 * q) * N_OUT];
            r[2 * q + 1] = w[(size_t)(16 * q + 1) * N_OUT];
        }
    };
    auto stage_store = [&](int buf, const float r[8]) {
        _Float16* d = &Bt[buf][sCol * 64 + 2 * u];
#pragma unroll
        for (int q = 0; q < 4; ++q) {
            v2h p;
            p.x = (_Float16)(colMask * r[2 * q]);
            p.y = (_Float16)(colMask * r[2 * q + 1]);
            *(v2h*)(d + 16 * q) = p;      // rows 16q+2u, 16q+2u+1
        }
    };

    float rg[8];
    stage_load(wRow, rg);                 // step 0 loads
    stage_store(0, rg);                   // fill buffer 0
    stage_load(wRow + stepStride, rg);    // step 1 loads in flight

    for (int step = 0; step < K_STEPS; ++step) {
        __syncthreads();
        if (step + 1 < K_STEPS) stage_store((step + 1) & 1, rg);
        if (step + 2 < K_STEPS)
            stage_load(wRow + (size_t)(step + 2) * stepStride, rg);
        if ((step + 4 < K_STEPS) && (sCol == 0)) {   // stream prefetch (+4)
            const float* pf = wRow + (size_t)(step + 4) * stepStride;
#pragma unroll
            for (int q = 0; q < 4; ++q) {
                __builtin_prefetch(pf + (size_t)(16 * q) * N_OUT, 0, 0);
                __builtin_prefetch(pf + (size_t)(16 * q + 1) * N_OUT, 0, 0);
            }
        }

#pragma unroll
        for (int j = 0; j < 2; ++j) {     // two K=32 WMMAs per barrier
            union { v16h v; v8h h[2]; } af, bf;
            const _Float16* ap = aPtr + j * 32;
            af.h[0] = *(const v8h*)(ap);          // K = [8g, 8g+8)
            af.h[1] = *(const v8h*)(ap + 16);     // K = [16+8g, 16+8g+8)
            const _Float16* bp = &Bt[step & 1][col * 64 + j * 32 + g * 8];
            bf.h[0] = *(const v8h*)(bp);          // K = [8g, 8g+8)
            bf.h[1] = *(const v8h*)(bp + 16);     // K = [16+8g, 16+8g+8)
            acc = __builtin_amdgcn_wmma_f32_16x16x32_f16(
                false, af.v, false, bf.v, (short)0, acc, false, false);
        }
        aPtr += 64;
    }

    // C layout: VGPR r, lanes 0-15 -> M=r, lanes 16-31 -> M=r+8; N = col
#pragma unroll
    for (int r = 0; r < 8; ++r) {
        int m = wave * 16 + r + g * 8;
        int n = n0 + col;
        if (n < N_OUT) atomicAdd(&out[(size_t)m * N_OUT + n], acc[r]);
    }
}

// ---------------------------------------------------------------------------
extern "C" void kernel_launch(void* const* d_in, const int* in_sizes, int n_in,
                              void* d_out, int out_size, void* d_ws, size_t ws_size,
                              hipStream_t stream) {
    (void)in_sizes; (void)n_in; (void)out_size; (void)ws_size;
    const float* topview = (const float*)d_in[0];
    const float* search  = (const float*)d_in[1];
    const float* w1      = (const float*)d_in[2];
    const float* b1      = (const float*)d_in[3];
    const float* w2      = (const float*)d_in[4];
    const float* b2      = (const float*)d_in[5];
    float* out = (float*)d_out;

    float*    coords = (float*)d_ws;                                     // 1 MB
    _Float16* A = (_Float16*)((char*)d_ws + (size_t)B_BATCH * H1 * 4);   // 12.6 MB

    coords_kernel<<<B_BATCH, 256, 0, stream>>>(topview, w1, b1, coords);
    gather_kernel<<<(B_BATCH * NPTS * 16) / 256, 256, 0, stream>>>(search, coords, A);
    init_out_kernel<<<(B_BATCH * N_OUT + 255) / 256, 256, 0, stream>>>(b2, out);
    gemm_kernel<<<dim3(63, K_CHUNKS), 128, 0, stream>>>(A, w2, out);
}
